// VectorQuantizerEMA_83537113907801
// MI455X (gfx1250) — matrix operand order repaired and checked by reading, not compiled, not run on hardware
//
#include <hip/hip_runtime.h>

typedef __attribute__((ext_vector_type(16))) _Float16 v16h;
typedef __attribute__((ext_vector_type(8)))  _Float16 v8h;
typedef __attribute__((ext_vector_type(8)))  float    v8f;
typedef __attribute__((ext_vector_type(4)))  float    v4f;

#define K_EMB 512
#define DIM 64
#define N_ROWS (64 * 4096)
#define WROW 72   /* 64 + 8 halves pad: row stride 144B, conflict-free-ish */
#define XROW 68   /* 64 + 4 floats pad: keeps float4 alignment per row   */

union B16u { v16h v; v8h h[2]; };

__global__ __launch_bounds__(256) void vq_main_kernel(
    const float* __restrict__ x, const float* __restrict__ w,
    float* __restrict__ out, float* __restrict__ gsse,
    float* __restrict__ gcounts)
{
    __shared__ _Float16 sW[K_EMB * WROW];     // 72 KB  f16 codebook
    __shared__ float    sWn[K_EMB];           //  2 KB  ||w_k||^2
    __shared__ float    sX[8 * 16 * XROW];    // 34 KB  x tiles (8 waves x 16 rows)
    __shared__ int      sIdx[8][16];

    const int tid  = threadIdx.x;
    const int lane = tid & 31;
    const int wv   = tid >> 5;
    const int hi   = lane >> 4;      // half-wave select
    const int m    = lane & 15;

    // ---- stage codebook f32 -> f16 into LDS (coalesced float4 reads) ----
    #pragma unroll
    for (int it = 0; it < 32; ++it) {
        int c = it * 256 + tid;          // float4 chunk 0..8191
        int k = c >> 4;
        int d = (c & 15) << 2;
        v4f v = *(const v4f*)(w + k * DIM + d);
        _Float16* dst = &sW[k * WROW + d];
        dst[0] = (_Float16)v.x; dst[1] = (_Float16)v.y;
        dst[2] = (_Float16)v.z; dst[3] = (_Float16)v.w;
    }
    // ---- codebook squared norms (full f32 precision, from global) ----
    #pragma unroll
    for (int rr = 0; rr < 2; ++rr) {
        int k = tid + rr * 256;
        float s = 0.f;
        #pragma unroll
        for (int d = 0; d < DIM; d += 4) {
            v4f v = *(const v4f*)(w + k * DIM + d);
            s += v.x * v.x + v.y * v.y + v.z * v.z + v.w * v.w;
        }
        sWn[k] = s;
    }
    // ---- stage this wave's 16x64 x-tile (coalesced float4) ----
    const size_t rowbase = (size_t)blockIdx.x * 128u + (size_t)wv * 16u;
    float* myX = &sX[wv * 16 * XROW];
    #pragma unroll
    for (int it = 0; it < 8; ++it) {
        int c = it * 32 + lane;          // chunk 0..255 within tile
        int r = c >> 4;
        int d = (c & 15) << 2;
        v4f v = *(const v4f*)(x + (rowbase + (size_t)r) * DIM + d);
        *(v4f*)&myX[r * XROW + d] = v;   // (68*r + d)*4 bytes: 16B aligned
    }
    __syncthreads();

    // ---- A fragments: 16x32 f16, ISA layout k = e + 8*(e>>3) + 8*hi ----
    v16h A0, A1;
    #pragma unroll
    for (int e = 0; e < 16; ++e) {
        int k = e + ((e >> 3) << 3) + (hi << 3);   // 0..31
        A0[e] = (_Float16)myX[m * XROW + k];
        A1[e] = (_Float16)myX[m * XROW + 32 + k];
    }

    float bestd[8];
    int   besti[8];
    #pragma unroll
    for (int r = 0; r < 8; ++r) { bestd[r] = 3.4e38f; besti[r] = 0; }

    // ---- 512 codes = 32 groups of 16; dist = ||w||^2 - 2*x.w ----
    #pragma unroll 4
    for (int kt = 0; kt < 32; ++kt) {
        int code = kt * 16 + m;          // this lane's column/code
        const _Float16* bp = &sW[code * WROW + (hi << 3)];
        B16u b0, b1;                     // B: 32x16, lane holds column, k = e + 8*(e>>3) + 8*hi
        b0.h[0] = *(const v8h*)(bp);
        b0.h[1] = *(const v8h*)(bp + 16);
        b1.h[0] = *(const v8h*)(bp + 32);
        b1.h[1] = *(const v8h*)(bp + 48);

        v8f acc = {};
        acc = __builtin_amdgcn_wmma_f32_16x16x32_f16(false, A0, false, b0.v,
                                                     (short)0, acc, false, false);
        acc = __builtin_amdgcn_wmma_f32_16x16x32_f16(false, A1, false, b1.v,
                                                     (short)0, acc, false, false);
        float wn = sWn[code];
        #pragma unroll
        for (int r = 0; r < 8; ++r) {            // row M = r + 8*hi, col = code
            float dist = wn - 2.0f * acc[r];
            if (dist < bestd[r]) { bestd[r] = dist; besti[r] = code; }
        }
    }

    // ---- argmin across the 16 lanes sharing each row (tie -> lower idx) ----
    #pragma unroll
    for (int r = 0; r < 8; ++r) {
        float dv = bestd[r]; int iv = besti[r];
        #pragma unroll
        for (int off = 8; off >= 1; off >>= 1) {
            float od = __shfl_xor(dv, off, 32);
            int   oi = __shfl_xor(iv, off, 32);
            if (od < dv || (od == dv && oi < iv)) { dv = od; iv = oi; }
        }
        if (m == 0) sIdx[wv][r + (hi << 3)] = iv;
    }
    __syncthreads();

    // ---- write quantized rows (exact f32 codebook), histogram + SSE ----
    float lsse = 0.f;
    #pragma unroll 4
    for (int r = 0; r < 16; ++r) {
        int idx = sIdx[wv][r];
        if (lane == 0) atomicAdd(&gcounts[idx], 1.0f);
        const float* wr = w + (size_t)idx * DIM;
        size_t orow = (rowbase + (size_t)r) * DIM;
        float q0 = wr[lane];
        float q1 = wr[lane + 32];
        out[orow + lane]      = q0;
        out[orow + lane + 32] = q1;
        float d0 = q0 - myX[r * XROW + lane];
        float d1 = q1 - myX[r * XROW + lane + 32];
        lsse += d0 * d0 + d1 * d1;
    }
    #pragma unroll
    for (int off = 16; off >= 1; off >>= 1) lsse += __shfl_xor(lsse, off, 32);
    if (lane == 0) atomicAdd(gsse, lsse);
}

__global__ __launch_bounds__(512) void vq_finalize_kernel(
    const float* __restrict__ gsse, const float* __restrict__ gcounts,
    const float* __restrict__ prior, float* __restrict__ out_loss)
{
    __shared__ float red[512];
    int t = threadIdx.x;
    float p = gcounts[t] * (1.0f / (float)N_ROWS);
    red[t] = p * (logf(p + 1e-10f) - logf(prior[t] + 1e-10f));
    __syncthreads();
    for (int s = 256; s > 0; s >>= 1) {
        if (t < s) red[t] += red[t + s];
        __syncthreads();
    }
    if (t == 0) {
        float mse = gsse[0] * (1.0f / ((float)N_ROWS * (float)DIM));
        out_loss[0] = 1.25f * mse + red[0];   // (q_latent + 0.25*e_latent) + KL
    }
}

extern "C" void kernel_launch(void* const* d_in, const int* in_sizes, int n_in,
                              void* d_out, int out_size, void* d_ws, size_t ws_size,
                              hipStream_t stream) {
    (void)in_sizes; (void)n_in; (void)out_size; (void)ws_size;
    const float* x     = (const float*)d_in[0];
    const float* w     = (const float*)d_in[1];
    const float* prior = (const float*)d_in[2];
    float* out = (float*)d_out;
    float* ws  = (float*)d_ws;
    float* gsse    = ws;       // [0]
    float* gcounts = ws + 1;   // [1..512]

    hipMemsetAsync(d_ws, 0, (1 + K_EMB) * sizeof(float), stream);
    vq_main_kernel<<<N_ROWS / 128, 256, 0, stream>>>(x, w, out, gsse, gcounts);
    vq_finalize_kernel<<<1, 512, 0, stream>>>(gsse, gcounts, prior,
                                              out + (size_t)N_ROWS * DIM);
}
